// AttentionLayer_29137058136845
// MI455X (gfx1250) — matrix operand order repaired and checked by reading, not compiled, not run on hardware
//
#include <hip/hip_runtime.h>

// AttentionLayer for MI455X (gfx1250, wave32, WMMA + TDM).
// B=2, S=4096, E=512, H=8, DH=64. scores scaled by 1/H (=1/8), softmax, P@V.
// Path: fp32 -> bf16 WMMA (v_wmma_f32_16x16x32_bf16) with fp32 accumulation.
// K/V tiles staged into LDS by the Tensor Data Mover (tensor_load_to_lds,
// s_wait_tensorcnt); all 4 waves of the block consume them via ds_load_b128.
// d_ws usage: 24 MB (Q,K bf16 [B,H,S,DH]; V bf16 transposed [B,H,DH,S]).

#define BB 2
#define SS 4096
#define EE 512
#define HH 8
#define DH 64

typedef __attribute__((ext_vector_type(16))) __bf16        v16bf;
typedef __attribute__((ext_vector_type(8)))  float         v8f;
typedef __attribute__((ext_vector_type(4)))  unsigned int  u32x4;
typedef __attribute__((ext_vector_type(2)))  unsigned int  u32x2;
typedef __attribute__((ext_vector_type(4)))  float         f32x4;
typedef __attribute__((ext_vector_type(4)))  unsigned int  tdm_g0_t;
typedef __attribute__((ext_vector_type(8)))  int           tdm_g1_t;
typedef __attribute__((ext_vector_type(4)))  int           tdm_g2_t;

union Frag16 {              // 16x32 bf16 operand fragment (8 VGPRs / lane)
    v16bf bf;
    u32x4 q[2];
    unsigned int   u[8];
    unsigned short us[16];
};

// Native bf16 converts (gfx1250 has full bf16): let the backend pick v_cvt ops
__device__ __forceinline__ unsigned short f2bf1(float a) {
    union { __bf16 h; unsigned short u; } c;
    c.h = (__bf16)a;
    return c.u;
}
__device__ __forceinline__ unsigned int f2bf_pk(float a, float b) {
    union { __bf16 h[2]; unsigned int u; } c;
    c.h[0] = (__bf16)a;
    c.h[1] = (__bf16)b;
    return c.u;
}

// ---------------------------------------------------------------------------
// TDM: issue a 2D tile copy Global -> LDS via the Tensor Data Mover.
// D# built per CDNA5 ISA ch.8: group0 {count=1 | lds_addr | global_addr |
// type=2}, group1 {data_size=1 (2B elems), tensor_dim0/1, tile_dim0/1,
// tensor_dim0_stride}, groups 2/3 zero (<=2D tensor). All units = elements.
// This toolchain exposes the 6-arg builtin (extra int32x8 group before cpol).
// ---------------------------------------------------------------------------
__device__ __forceinline__ void tdm_load_2d(
    unsigned lds_addr, const void* gaddr,
    unsigned tensor_d0, unsigned tensor_d1,
    unsigned tile_d0, unsigned tile_d1,
    unsigned stride0)
{
    unsigned long long ga = (unsigned long long)(size_t)gaddr;
    tdm_g0_t g0 = {
        1u,                                                   // count=1, user
        lds_addr,                                             // LDS byte addr
        (unsigned)ga,                                         // gaddr[31:0]
        (unsigned)((ga >> 32) & 0x01FFFFFFull) | (2u << 30)   // gaddr[56:32], type=2
    };
    tdm_g1_t g1 = {
        (int)(1u << 16),                                      // data_size=2B, mask=0
        (int)(tensor_d0 << 16),                               // tensor_dim0[15:0]
        (int)((tensor_d0 >> 16) | (tensor_d1 << 16)),         // dim0 hi | dim1 lo
        (int)((tensor_d1 >> 16) | (tile_d0 << 16)),           // dim1 hi | tile_dim0
        (int)(tile_d1 & 0xFFFFu),                             // tile_dim1 (dim2=0)
        (int)stride0,                                         // dim0_stride[31:0]
        0, 0                                                  // stride hi / dim1_stride
    };
    tdm_g2_t z4 = {0, 0, 0, 0};
    tdm_g1_t z8 = {0, 0, 0, 0, 0, 0, 0, 0};
    __builtin_amdgcn_tensor_load_to_lds(g0, g1, z4, z4, z8, 0);
}

// ---------------------------------------------------------------------------
// Kernel 1: fused QKV projection GEMM.  grid = (M/64, E/64, 3), block = 128.
// A tile (X) and B tile (W, stored transposed) staged in LDS as bf16.
// Outputs: z=0 -> Q [B,H,S,DH], z=1 -> K [B,H,S,DH], z=2 -> V^T [B,H,DH,S].
// ---------------------------------------------------------------------------
__global__ __launch_bounds__(128) void qkv_proj_wmma(
    const float* __restrict__ X,
    const float* __restrict__ Wq,
    const float* __restrict__ Wk,
    const float* __restrict__ Wv,
    unsigned short* __restrict__ Qb,
    unsigned short* __restrict__ Kb,
    unsigned short* __restrict__ Vb)
{
    __shared__ unsigned short Xs[64 * 32];   // 64 rows x 32 k, bf16
    __shared__ unsigned short Ws[64 * 32];   // Ws[n][k] = W[k0+k][n0+n], bf16

    const int t  = threadIdx.x;
    const int wv = t >> 5;
    const int ln = t & 31;
    const int lh = ln & 15;
    const bool hi = (ln >= 16);

    const int m0 = blockIdx.x * 64;          // rows of X (b*S + s)
    const int n0 = blockIdx.y * 64;          // output feature
    const int z  = blockIdx.z;               // 0=Q, 1=K, 2=V
    const float* W = (z == 0) ? Wq : ((z == 1) ? Wk : Wv);

    v8f acc[4];
#pragma unroll
    for (int g = 0; g < 4; ++g) acc[g] = (v8f){0.f,0.f,0.f,0.f,0.f,0.f,0.f,0.f};

    for (int k0 = 0; k0 < EE; k0 += 32) {
        // ---- stage X tile: 64x32 fp32 -> bf16 (vectorized b128 loads) ----
#pragma unroll
        for (int c = 0; c < 4; ++c) {
            int linear = (t + c * 128) * 4;                 // 0..2044
            int row = linear >> 5, col = linear & 31;
            f32x4 xv = *(const f32x4*)(X + (m0 + row) * EE + k0 + col);
            u32x2 st = {f2bf_pk(xv.x, xv.y), f2bf_pk(xv.z, xv.w)};
            *(u32x2*)(&Xs[row * 32 + col]) = st;
        }
        // ---- stage W tile transposed: Ws[n][k] = W[k0+k][n0+n] ----
#pragma unroll
        for (int j = 0; j < 16; ++j) {
            int linear = t + j * 128;                        // 0..2047
            int n = linear & 63, k = linear >> 6;            // coalesced over n
            Ws[n * 32 + k] = f2bf1(W[(k0 + k) * EE + n0 + n]);
        }
        __syncthreads();

        // ---- A fragment: row = wave's 16-row slice, ISA 16-bit A layout ----
        Frag16 a;
        {
            int row = wv * 16 + lh;
            int c0  = hi ? 8 : 0;                            // K 8..15 / 0..7
            a.q[0] = *(const u32x4*)(&Xs[row * 32 + c0]);
            a.q[1] = *(const u32x4*)(&Xs[row * 32 + c0 + 16]);
        }
        // ---- 4 B fragments (16 output cols each) + WMMA ----
#pragma unroll
        for (int g = 0; g < 4; ++g) {
            Frag16 bfr;
            int n    = g * 16 + lh;
            int koff = hi ? 8 : 0;                           // K 8..15 / 0..7
            bfr.q[0] = *(const u32x4*)(&Ws[n * 32 + koff]);
            bfr.q[1] = *(const u32x4*)(&Ws[n * 32 + koff + 16]);
            acc[g] = __builtin_amdgcn_wmma_f32_16x16x32_bf16(
                false, a.bf, false, bfr.bf, (short)0, acc[g], false, false);
        }
        __syncthreads();
    }

    // ---- epilogue: bf16 stores into attention layouts ----
#pragma unroll
    for (int g = 0; g < 4; ++g) {
#pragma unroll
        for (int i = 0; i < 8; ++i) {
            int r = (hi ? 8 : 0) + i;                        // C-layout row
            int m = m0 + wv * 16 + r;
            int e = n0 + g * 16 + lh;
            int b = m >> 12, s = m & (SS - 1);
            int h = e >> 6, d = e & (DH - 1);
            unsigned short val = f2bf1(acc[g][i]);
            if (z == 2) {
                Vb[(((b << 3) + h) * DH + d) * SS + s] = val;    // V^T
            } else {
                unsigned short* O = (z == 0) ? Qb : Kb;
                O[(((b << 3) + h) * SS + s) * DH + d] = val;
            }
        }
    }
}

// ---------------------------------------------------------------------------
// Kernel 2: flash attention.  grid = (S/64, B*H), block = 128 (4 waves).
// Each wave owns 16 query rows; block streams a shared 32-key tile per
// iteration.  Wave 0 drives the TDM (tensor_load_to_lds K + V^T tiles,
// s_wait_tensorcnt), wave 1 prefetches the next tile toward L2; all waves do
// 4 WMMA (Q.K^T) + online softmax (half-wave shuffles) + 4 WMMA (P.V).
// ---------------------------------------------------------------------------
__global__ __launch_bounds__(128) void flash_attn_wmma(
    const unsigned short* __restrict__ Qg,
    const unsigned short* __restrict__ Kg,
    const unsigned short* __restrict__ Vtg,   // [B,H,DH,S]
    float* __restrict__ out)
{
    __shared__ unsigned short Ks[32 * DH];      // [key][dh]   4 KB
    __shared__ unsigned short Vs[DH * 32];      // [dh][key]   4 KB
    __shared__ unsigned short Pb[4][16 * 32];   // per-wave P staging, 4 KB

    const int t  = threadIdx.x;
    const int wv = t >> 5;
    const int ln = t & 31;
    const int lh = ln & 15;
    const bool hi = (ln >= 16);

    const int bh = blockIdx.y;
    const int b = bh >> 3, h = bh & 7;
    const int s0 = blockIdx.x * 64 + wv * 16;

    const unsigned short* Qb = Qg + ((long)bh * SS + s0) * DH;
    const unsigned short* Kb = Kg + (long)bh * SS * DH;
    const unsigned short* Vb = Vtg + (long)bh * DH * SS;

    // Q A-fragments for k-depth 0..31 and 32..63 (persist across the loop)
    Frag16 aq[2];
    {
        int c0 = hi ? 8 : 0;
#pragma unroll
        for (int f = 0; f < 2; ++f) {
            const unsigned short* p = Qb + lh * DH + f * 32 + c0;
            aq[f].q[0] = *(const u32x4*)p;
            aq[f].q[1] = *(const u32x4*)(p + 16);
        }
    }

    float mrow[8], lrow[8];
#pragma unroll
    for (int i = 0; i < 8; ++i) { mrow[i] = -3.0e38f; lrow[i] = 0.f; }
    v8f o[4];
#pragma unroll
    for (int g = 0; g < 4; ++g) o[g] = (v8f){0.f,0.f,0.f,0.f,0.f,0.f,0.f,0.f};

    const float LOG2E = 1.44269504088896f;
    const unsigned ldsK = (unsigned)(size_t)(void*)Ks;
    const unsigned ldsV = (unsigned)(size_t)(void*)Vs;

    for (int t0 = 0; t0 < SS; t0 += 32) {
        __syncthreads();                 // previous tile fully consumed
        if (wv == 0) {
            // K tile: 32 keys x 64 dh  (row-major [key][dh], stride DH)
            tdm_load_2d(ldsK, Kb + t0 * DH, DH, SS, DH, 32, DH);
            // V tile: 64 dh x 32 keys from V^T (row-major [dh][s], stride S)
            tdm_load_2d(ldsV, Vb + t0, SS, DH, 32, DH, SS);
            __builtin_amdgcn_s_wait_tensorcnt(0);
        } else if (wv == 1 && t0 + 32 < SS) {
            // pull next tile's lines toward L2 while this tile is computed
            __builtin_prefetch(Kb + (t0 + 32 + ln) * DH, 0, 0);
            __builtin_prefetch(Vb + ln * SS + t0 + 32, 0, 0);
            __builtin_prefetch(Vb + (32 + ln) * SS + t0 + 32, 0, 0);
        }
        __syncthreads();                 // tile ready for all waves

        // ---- scores: S = (Q.K^T) * (1/H) for 2 groups of 16 keys ----
        v8f sc[2];
#pragma unroll
        for (int g = 0; g < 2; ++g) {
            v8f acc = (v8f){0.f,0.f,0.f,0.f,0.f,0.f,0.f,0.f};
            int lkey = g * 16 + lh;
#pragma unroll
            for (int f = 0; f < 2; ++f) {
                Frag16 bk;
                const unsigned short* p = &Ks[lkey * DH + f * 32 + (hi ? 8 : 0)];
                bk.q[0] = *(const u32x4*)p;
                bk.q[1] = *(const u32x4*)(p + 16);
                acc = __builtin_amdgcn_wmma_f32_16x16x32_bf16(
                    false, aq[f].bf, false, bk.bf, (short)0, acc, false, false);
            }
            sc[g] = acc * 0.125f;              // scale = 1/N_HEADS
        }

        // ---- online softmax: row stats via half-wave xor shuffles ----
        float fac[8];
#pragma unroll
        for (int i = 0; i < 8; ++i) {
            float mt = fmaxf(sc[0][i], sc[1][i]);
            mt = fmaxf(mt, __shfl_xor(mt, 1, 32));
            mt = fmaxf(mt, __shfl_xor(mt, 2, 32));
            mt = fmaxf(mt, __shfl_xor(mt, 4, 32));
            mt = fmaxf(mt, __shfl_xor(mt, 8, 32));
            float mn = fmaxf(mrow[i], mt);
            float p0 = __builtin_amdgcn_exp2f((sc[0][i] - mn) * LOG2E);
            float p1 = __builtin_amdgcn_exp2f((sc[1][i] - mn) * LOG2E);
            float fr = __builtin_amdgcn_exp2f((mrow[i] - mn) * LOG2E);
            float rs = p0 + p1;
            rs += __shfl_xor(rs, 1, 32);
            rs += __shfl_xor(rs, 2, 32);
            rs += __shfl_xor(rs, 4, 32);
            rs += __shfl_xor(rs, 8, 32);
            lrow[i] = lrow[i] * fr + rs;
            mrow[i] = mn;
            fac[i]  = fr;
            sc[0][i] = p0;
            sc[1][i] = p1;
        }
#pragma unroll
        for (int g = 0; g < 4; ++g)
#pragma unroll
            for (int i = 0; i < 8; ++i) o[g][i] *= fac[i];

        // ---- C-layout -> A-layout reshape of P through per-wave LDS ----
        unsigned short* pb = &Pb[wv][0];
#pragma unroll
        for (int g = 0; g < 2; ++g)
#pragma unroll
            for (int i = 0; i < 8; ++i) {
                int r = (hi ? 8 : 0) + i;
                int c = g * 16 + lh;
                pb[r * 32 + c] = f2bf1(sc[g][i]);
            }
        __builtin_amdgcn_wave_barrier();       // DS ops are in-order per wave
        Frag16 pa;
        {
            int c0 = hi ? 8 : 0;
            pa.q[0] = *(const u32x4*)(&pb[lh * 32 + c0]);
            pa.q[1] = *(const u32x4*)(&pb[lh * 32 + c0 + 16]);
        }
        __builtin_amdgcn_wave_barrier();

        // ---- O += P.V  (B frags from the LDS V tile) ----
#pragma unroll
        for (int g = 0; g < 4; ++g) {
            Frag16 bv;
            int d = g * 16 + lh;
            const unsigned short* p = &Vs[d * 32 + (hi ? 8 : 0)];
            bv.q[0] = *(const u32x4*)p;
            bv.q[1] = *(const u32x4*)(p + 16);
            o[g] = __builtin_amdgcn_wmma_f32_16x16x32_bf16(
                false, pa.bf, false, bv.bf, (short)0, o[g], false, false);
        }
    }

    // ---- normalize + write fp32 output [B,S,E] ----
#pragma unroll
    for (int g = 0; g < 4; ++g)
#pragma unroll
        for (int i = 0; i < 8; ++i) {
            int r = (hi ? 8 : 0) + i;
            int s = s0 + r;
            int e = h * DH + g * 16 + lh;
            out[((long)b * SS + s) * EE + e] = o[g][i] / lrow[i];
        }
}

// ---------------------------------------------------------------------------
extern "C" void kernel_launch(void* const* d_in, const int* in_sizes, int n_in,
                              void* d_out, int out_size, void* d_ws, size_t ws_size,
                              hipStream_t stream) {
    const float* X  = (const float*)d_in[0];
    const float* Wq = (const float*)d_in[1];
    const float* Wk = (const float*)d_in[2];
    const float* Wv = (const float*)d_in[3];

    // Workspace partition (bf16): Q,K [B,H,S,DH], V^T [B,H,DH,S] = 24 MB total
    unsigned short* qb = (unsigned short*)d_ws;
    unsigned short* kb = qb + (size_t)BB * SS * EE;
    unsigned short* vb = kb + (size_t)BB * SS * EE;

    dim3 g1((BB * SS) / 64, EE / 64, 3);
    qkv_proj_wmma<<<g1, dim3(128), 0, stream>>>(X, Wq, Wk, Wv, qb, kb, vb);

    dim3 g2(SS / 64, BB * HH);
    flash_attn_wmma<<<g2, dim3(128), 0, stream>>>(qb, kb, vb, (float*)d_out);
}
